// BOW_22728966931075
// MI455X (gfx1250) — matrix operand order repaired and checked by reading, not compiled, not run on hardware
//
#include <hip/hip_runtime.h>

#define VOCAB   100000
#define EMB     128
#define CLASSES 1000
#define NPAD    1024        // 64 tiles of 16 -> 16 groups of 4; rows >= 1000 zero
#define BATCH   16384
#define SEQLEN  200

typedef __bf16 bf16;
typedef __attribute__((ext_vector_type(4)))  __bf16 bf16x4;
typedef __attribute__((ext_vector_type(8)))  __bf16 bf16x8;
typedef __attribute__((ext_vector_type(16))) __bf16 bf16x16;
typedef __attribute__((ext_vector_type(4)))  float  v4f;
typedef __attribute__((ext_vector_type(8)))  float  v8f;

// ---------------- Kernel 1: fc_w f32 -> bf16, padded to NPAD rows ----------
__global__ void cvt_w_kernel(const float* __restrict__ w, bf16* __restrict__ wq) {
    int idx = blockIdx.x * 256 + threadIdx.x;      // [0, NPAD*EMB)
    int n = idx >> 7;                               // row (class)
    int k = idx & 127;                              // column (emb dim)
    float v = (n < CLASSES) ? w[n * EMB + k] : 0.0f;
    wq[idx] = (bf16)v;
}

// ---------------- Kernel 2: masked embedding-bag (sum pool) ----------------
// One wave32 per batch row: 32 lanes x float4 = 128 dims. Token id is made
// wave-uniform via readfirstlane so the padding branch is scalar and the
// embedding row load is a coalesced 512B burst out of the L2-resident table.
__global__ void embed_pool_kernel(const int* __restrict__ seq,
                                  const float* __restrict__ emb,
                                  bf16* __restrict__ pooled) {
    int row  = (blockIdx.x * blockDim.x + threadIdx.x) >> 5;   // batch row
    int lane = threadIdx.x & 31;
    const int* s = seq + (long)row * SEQLEN;

    v4f acc = {0.f, 0.f, 0.f, 0.f};
    for (int l = 0; l < SEQLEN; ++l) {
        int tok = __builtin_amdgcn_readfirstlane(s[l]);
        if (tok != 0) {   // id 0 = padding, excluded from the bag
            const v4f v = *(const v4f*)(emb + (long)tok * EMB + lane * 4);
            acc += v;
        }
    }
    bf16x4 o;
    o[0] = (bf16)acc[0]; o[1] = (bf16)acc[1];
    o[2] = (bf16)acc[2]; o[3] = (bf16)acc[3];
    *(bf16x4*)(pooled + (long)row * EMB + lane * 4) = o;
}

// ---------------- Kernel 3: [16384,128] x [128,1024] via WMMA bf16 ---------
// One wave per 16(M) x 64(N) output block: 4 accumulators, each A fragment
// reused by 4 consecutive N-tiles -> 16 x v_wmma_f32_16x16x32_bf16 per wave
// and 4x less A traffic from L2 than a 16x16-per-wave scheme.
// A feed (contiguous k order, same as B): lane holds K[k0..k0+8) and K[k0+8..k0+16),
//   k0 = (lane<16 ? 0 : 16), row M = lane%16  -> two b128 loads + shuffle.
// B feed (contiguous k order, same as A): lane holds 16 contiguous K at
//   (lane<16 ? 0 : 16), column N = lane%16   -> one 32B load; fc_w is [N,K]
//   row-major so no transpose is needed.
__global__ void gemm_wmma_kernel(const bf16* __restrict__ A,     // [BATCH, EMB]
                                 const bf16* __restrict__ Wq,    // [NPAD, EMB]
                                 const float* __restrict__ bias, // [CLASSES]
                                 float* __restrict__ out) {      // [BATCH, CLASSES]
    const int NG  = NPAD / 64;                                   // 16 N-groups
    int wid  = (blockIdx.x * blockDim.x + threadIdx.x) >> 5;     // global wave id
    int lane = threadIdx.x & 31;
    int ng = wid % NG;
    int mt = wid / NG;
    int m0 = mt * 16;
    int n0 = ng * 64;
    int half = lane >> 4;      // 0: lanes 0-15, 1: lanes 16-31
    int l16  = lane & 15;

    const bf16* ap = A  + (long)(m0 + l16) * EMB + half * 16;
    const bf16* bp = Wq + (long)(n0 + l16) * EMB + half * 16;

    v8f c0 = {}, c1 = {}, c2 = {}, c3 = {};
#pragma unroll
    for (int k = 0; k < EMB; k += 32) {
        bf16x8 alo = *(const bf16x8*)(ap + k);        // K[k0 .. k0+8)
        bf16x8 ahi = *(const bf16x8*)(ap + k + 8);    // K[k0+8 .. k0+16)
        bf16x16 a = __builtin_shufflevector(alo, ahi,
                     0, 1, 2, 3, 4, 5, 6, 7, 8, 9, 10, 11, 12, 13, 14, 15);
        bf16x16 b0 = *(const bf16x16*)(bp + k);                 // N-tile 0
        bf16x16 b1 = *(const bf16x16*)(bp + k + 16 * EMB);      // N-tile 1
        bf16x16 b2 = *(const bf16x16*)(bp + k + 32 * EMB);      // N-tile 2
        bf16x16 b3 = *(const bf16x16*)(bp + k + 48 * EMB);      // N-tile 3
        c0 = __builtin_amdgcn_wmma_f32_16x16x32_bf16(false, a, false, b0, (short)0, c0, false, false);
        c1 = __builtin_amdgcn_wmma_f32_16x16x32_bf16(false, a, false, b1, (short)0, c1, false, false);
        c2 = __builtin_amdgcn_wmma_f32_16x16x32_bf16(false, a, false, b2, (short)0, c2, false, false);
        c3 = __builtin_amdgcn_wmma_f32_16x16x32_bf16(false, a, false, b3, (short)0, c3, false, false);
    }

    // C/D layout: VGPR i -> M = i + 8*half, N = lane%16
    int mbase = m0 + half * 8;
    v8f acc[4] = {c0, c1, c2, c3};
#pragma unroll
    for (int j = 0; j < 4; ++j) {
        int col = n0 + j * 16 + l16;
        if (col < CLASSES) {
            float bb = bias[col];
#pragma unroll
            for (int i = 0; i < 8; ++i)
                out[(long)(mbase + i) * CLASSES + col] = acc[j][i] + bb;
        }
    }
}

extern "C" void kernel_launch(void* const* d_in, const int* in_sizes, int n_in,
                              void* d_out, int out_size, void* d_ws, size_t ws_size,
                              hipStream_t stream) {
    const int*   seq = (const int*)  d_in[0];   // [BATCH, SEQLEN] int32
    const float* emb = (const float*)d_in[1];   // [VOCAB, EMB]    f32
    const float* fcw = (const float*)d_in[2];   // [CLASSES, EMB]  f32
    const float* fcb = (const float*)d_in[3];   // [CLASSES]       f32
    float* out = (float*)d_out;                 // [BATCH, CLASSES] f32

    // Workspace layout: [0, 262144): fc_w bf16 padded [1024,128];
    //                   [262144, +4MiB): pooled bf16 [16384,128]
    bf16* wq     = (bf16*)d_ws;
    bf16* pooled = (bf16*)((char*)d_ws + 262144);

    cvt_w_kernel<<<(NPAD * EMB) / 256, 256, 0, stream>>>(fcw, wq);
    embed_pool_kernel<<<BATCH / 8, 256, 0, stream>>>(seq, emb, pooled);  // 8 waves/block = 8 rows
    // 1024 M-tiles x 16 N-groups = 16384 waves / 8 waves per block
    gemm_wmma_kernel<<<(BATCH / 16) * (NPAD / 64) / 8, 256, 0, stream>>>(pooled, wq, fcb, out);
}
